// Intention_49967649522235
// MI455X (gfx1250) — compile-verified, matrix operand order checked by reading, not run on hardware
//
#include <hip/hip_runtime.h>
#include <hip/hip_bf16.h>

// MI455X (gfx1250, CDNA5, wave32). All matmuls routed through
// v_wmma_f32_16x16x32_bf16 (f32 accumulate); bf16 tile staging routed through
// the async global->LDS engines (ASYNCcnt) where no conversion is needed.

typedef __bf16 bf16;
typedef __attribute__((ext_vector_type(16))) __bf16 v16bf;
typedef __attribute__((ext_vector_type(8)))  float  v8f;

#define B_    4
#define S_    4096
#define E_    2048
#define NH_   16
#define HD_   128
#define MTOT  (B_ * S_)   // 16384

// ---- async global->LDS (gfx1250) -----------------------------------------
#if defined(__has_builtin)
#  if __has_builtin(__builtin_amdgcn_global_load_async_to_lds_b128)
#    define HAVE_ASYNC_LDS 1
#  endif
#endif
#ifndef HAVE_ASYNC_LDS
#  define HAVE_ASYNC_LDS 0
#endif

// Exact pointee type from the toolchain's builtin signature:
// 'int __attribute__((vector_size(16)))' in addrspace(1)/(3).
typedef int v4i_vs __attribute__((vector_size(16)));
typedef __attribute__((address_space(1))) v4i_vs* g_v4i_ptr;
typedef __attribute__((address_space(3))) v4i_vs* l_v4i_ptr;

// Copy 16 bytes (8 bf16) global -> LDS.  Async path when the toolchain
// exposes the gfx1250 builtin; plain vector copy otherwise.
__device__ __forceinline__ void copy16_g2l(const bf16* g, bf16* l) {
#if HAVE_ASYNC_LDS
  __builtin_amdgcn_global_load_async_to_lds_b128(
      (g_v4i_ptr)(bf16*)g, (l_v4i_ptr)l, /*offset=*/0, /*cpol=*/0);
#else
  *(uint4*)l = *(const uint4*)g;
#endif
}

__device__ __forceinline__ void async_join() {
#if HAVE_ASYNC_LDS
#  if __has_builtin(__builtin_amdgcn_s_wait_asynccnt)
  __builtin_amdgcn_s_wait_asynccnt(0);
#  else
  asm volatile("s_wait_asynccnt 0" ::: "memory");
#  endif
#endif
}

// ---------------------------------------------------------------------------
union FragU { unsigned int u[8]; v16bf v; };

__device__ __forceinline__ v8f zero_v8f() {
  v8f z;
#pragma unroll
  for (int i = 0; i < 8; ++i) z[i] = 0.0f;
  return z;
}

// K-pair start index held by VGPR v for lane-half hi16 (ISA 7.12.2, 16-bit A 16x32).
__device__ __forceinline__ int kpair_base(int v, int hi16) {
  return ((v & 4) << 2) | (hi16 << 3) | ((v & 3) << 1);  // 16*(v>=4) + 8*hi + 2*(v&3)
}

// Load a 16x32 bf16 fragment from an LDS tile with row stride RS (elements).
template<int RS>
__device__ __forceinline__ FragU load_frag(const bf16* tile, int row, int kofs, int lane) {
  FragU f;
  const unsigned int* p = (const unsigned int*)(tile + row * RS + kofs);
  const int hi16 = lane >> 4;
#pragma unroll
  for (int v = 0; v < 8; ++v) {
    int k = kpair_base(v, hi16);
    f.u[v] = p[k >> 1];
  }
  return f;
}

__device__ __forceinline__ v8f wmma_bf16(const FragU& a, const FragU& b, v8f c) {
  return __builtin_amdgcn_wmma_f32_16x16x32_bf16(
      /*neg_a=*/false, a.v, /*neg_b=*/false, b.v,
      /*c_mod=*/(short)0, c, /*reuse_a=*/false, /*reuse_b=*/false);
}

// ---------------------------------------------------------------------------
// GEMM: C[M,N] = A[M,K] @ B[K,N] + bias.  A is f32 or bf16; B/bias are f32.
// Block tile 128x128, K-step 32, 256 threads = 8 waves; each wave owns 64x32.
// bf16 A tiles are staged via async global->LDS copies.
// ---------------------------------------------------------------------------
template<bool A_F32, bool OUT_F32>
__global__ __launch_bounds__(256)
void gemm_bias_kernel(const void* __restrict__ Ain, const float* __restrict__ Bw,
                      const float* __restrict__ bias, void* __restrict__ Cout,
                      int M, int N, int K) {
  __shared__ __align__(16) bf16 As[128 * 32];   // A tile, row-major
  __shared__ __align__(16) bf16 Bt[128 * 32];   // B tile transposed: Bt[n][k]

  const int tid  = threadIdx.x;
  const int lane = tid & 31;
  const int wave = tid >> 5;
  const int wr   = wave >> 2;     // 0..1 : 64-row band
  const int wc   = wave & 3;      // 0..3 : 32-col band
  const int m0   = blockIdx.y * 128;
  const int n0   = blockIdx.x * 128;

  const float* Af = (const float*)Ain;
  const bf16*  Ab = (const bf16*)Ain;

  v8f acc[4][2];
#pragma unroll
  for (int i = 0; i < 4; ++i)
#pragma unroll
    for (int j = 0; j < 2; ++j) acc[i][j] = zero_v8f();

  for (int k0 = 0; k0 < K; k0 += 32) {
    if (A_F32) {
      // Stage A tile with f32 -> bf16 conversion (VALU path; async can't convert).
#pragma unroll
      for (int i = 0; i < 16; ++i) {
        int idx = tid + i * 256;           // 4096 elements
        int r = idx >> 5, c = idx & 31;
        As[r * 32 + c] = (bf16)Af[(size_t)(m0 + r) * K + (k0 + c)];
      }
    } else {
      // bf16 A tile: straight copy -> async global->LDS, 16B per lane issue.
#pragma unroll
      for (int i = 0; i < 2; ++i) {
        int ci = tid + i * 256;            // 512 chunks of 8 bf16
        int r = ci >> 2, c = (ci & 3) * 8;
        copy16_g2l(Ab + (size_t)(m0 + r) * K + (k0 + c), As + r * 32 + c);
      }
    }
    // Stage B tile transposed (f32 weights -> bf16).
#pragma unroll
    for (int i = 0; i < 16; ++i) {
      int idx = tid + i * 256;
      int kk = idx >> 7, n = idx & 127;
      Bt[n * 32 + kk] = (bf16)Bw[(size_t)(k0 + kk) * N + (n0 + n)];
    }
    // Prefetch next K-tile (global_prefetch_b8 on gfx1250).
    if (k0 + 32 < K) {
      const char* pa = (const char*)Ain +
          ((size_t)(m0 + (tid >> 1)) * K + (k0 + 32)) * (A_F32 ? 4 : 2);
      __builtin_prefetch(pa, 0, 1);
      const char* pb = (const char*)Bw +
          ((size_t)(k0 + 32 + (tid >> 7)) * N + (n0 + (tid & 127))) * 4;
      __builtin_prefetch(pb, 0, 1);
    }
    if (!A_F32) async_join();
    __syncthreads();

    FragU bfrag[2];
#pragma unroll
    for (int j = 0; j < 2; ++j)
      bfrag[j] = load_frag<32>(Bt, wc * 32 + j * 16 + (lane & 15), 0, lane);
#pragma unroll
    for (int i = 0; i < 4; ++i) {
      FragU afrag = load_frag<32>(As, wr * 64 + i * 16 + (lane & 15), 0, lane);
#pragma unroll
      for (int j = 0; j < 2; ++j)
        acc[i][j] = wmma_bf16(afrag, bfrag[j], acc[i][j]);
    }
    __syncthreads();
  }

  const int hi16 = lane >> 4;
  const int nc   = lane & 15;
#pragma unroll
  for (int i = 0; i < 4; ++i) {
#pragma unroll
    for (int j = 0; j < 2; ++j) {
      int colg = n0 + wc * 32 + j * 16 + nc;
      float bv = bias ? bias[colg] : 0.0f;
#pragma unroll
      for (int v = 0; v < 8; ++v) {
        int rowg = m0 + wr * 64 + i * 16 + v + hi16 * 8;
        float val = acc[i][j][v] + bv;
        if (OUT_F32) ((float*)Cout)[(size_t)rowg * N + colg] = val;
        else         ((bf16*) Cout)[(size_t)rowg * N + colg] = (bf16)val;
      }
    }
  }
}

// ---------------------------------------------------------------------------
// Per-head kk = K^T K + alpha*I  and  ktv = K^T V.  One WG per (b,h).
// K/V tiles staged transposed in LDS: Kt[d][s] serves both A (row d of K^T)
// and B (column e of K) fragments.  (Transposed store -> VALU staging.)
// ---------------------------------------------------------------------------
__global__ __launch_bounds__(256)
void kkt_ktv_kernel(const bf16* __restrict__ Kb, const bf16* __restrict__ Vb,
                    const float* __restrict__ alpha,
                    float* __restrict__ kk, float* __restrict__ ktv) {
  __shared__ __align__(16) bf16 Kt[128 * 32];
  __shared__ __align__(16) bf16 Vt[128 * 32];

  const int head = blockIdx.x;
  const int b = head / NH_, h = head % NH_;
  const size_t base = (size_t)b * S_ * E_ + (size_t)h * HD_;
  const int tid = threadIdx.x, lane = tid & 31, wave = tid >> 5;

  v8f accK[8], accV[8];
#pragma unroll
  for (int e = 0; e < 8; ++e) { accK[e] = zero_v8f(); accV[e] = zero_v8f(); }

  for (int s0 = 0; s0 < S_; s0 += 32) {
#pragma unroll
    for (int i = 0; i < 16; ++i) {
      int idx = tid + i * 256;
      int ss = idx >> 7, d = idx & 127;
      size_t g = base + (size_t)(s0 + ss) * E_ + d;
      Kt[d * 32 + ss] = Kb[g];
      Vt[d * 32 + ss] = Vb[g];
    }
    __syncthreads();
    FragU afrag = load_frag<32>(Kt, wave * 16 + (lane & 15), 0, lane);
#pragma unroll
    for (int e = 0; e < 8; ++e) {
      FragU bk = load_frag<32>(Kt, e * 16 + (lane & 15), 0, lane);
      accK[e] = wmma_bf16(afrag, bk, accK[e]);
      FragU bv = load_frag<32>(Vt, e * 16 + (lane & 15), 0, lane);
      accV[e] = wmma_bf16(afrag, bv, accV[e]);
    }
    __syncthreads();
  }

  const float a0 = alpha[0];
  const int hi16 = lane >> 4, nc = lane & 15;
  float* kkh = kk  + (size_t)head * HD_ * HD_;
  float* kvh = ktv + (size_t)head * HD_ * HD_;
#pragma unroll
  for (int e = 0; e < 8; ++e) {
    int col = e * 16 + nc;
#pragma unroll
    for (int v = 0; v < 8; ++v) {
      int row = wave * 16 + v + hi16 * 8;
      kkh[row * 128 + col] = accK[e][v] + (row == col ? a0 : 0.0f);
      kvh[row * 128 + col] = accV[e][v];
    }
  }
}

// ---------------------------------------------------------------------------
// In-place Gauss-Jordan inverse of a 128x128 SPD matrix in LDS (no pivoting).
// Thread t owns row t>>1, column half (t&1)*64.
// ---------------------------------------------------------------------------
__global__ __launch_bounds__(256)
void invert_kernel(float* __restrict__ kk) {
  __shared__ float A[128][128];   // 64 KB
  const int tid = threadIdx.x;
  float* Ag = kk + (size_t)blockIdx.x * 128 * 128;
  for (int i = 0; i < 64; ++i) {
    int idx = tid + i * 256;
    A[idx >> 7][idx & 127] = Ag[idx];
  }
  const int r  = tid >> 1;
  const int c0 = (tid & 1) * 64;
  for (int p = 0; p < 128; ++p) {
    __syncthreads();
    float inv = 1.0f / A[p][p];
    if (r == p) {
      for (int c = c0; c < c0 + 64; ++c)
        if (c != p) A[p][c] *= inv;
    }
    __syncthreads();
    float f = (r == p) ? 0.0f : A[r][p];
    __syncthreads();
    if (r != p) {
      for (int c = c0; c < c0 + 64; ++c)
        if (c != p) A[r][c] -= f * A[p][c];
      if (p >= c0 && p < c0 + 64) A[r][p] = -f * inv;
    } else {
      if (p >= c0 && p < c0 + 64) A[p][p] = inv;
    }
  }
  __syncthreads();
  for (int i = 0; i < 64; ++i) {
    int idx = tid + i * 256;
    Ag[idx] = A[idx >> 7][idx & 127];
  }
}

// ---------------------------------------------------------------------------
// score = softmax(kk_inv @ ktv) per head.  WMMA for the 128^3 matmul, then
// row softmax on the f32 product (LDS aliased via dynamic shared memory).
// ---------------------------------------------------------------------------
__global__ __launch_bounds__(256)
void score_kernel(const float* __restrict__ kkinv, const float* __restrict__ ktv,
                  bf16* __restrict__ score) {
  extern __shared__ char smem4[];           // 64 KB dynamic
  bf16*  smA = (bf16*)smem4;                // inv, row-major   (32 KB)
  bf16*  smB = (bf16*)(smem4 + 32768);      // ktv^T            (32 KB)
  float* smP = (float*)smem4;               // product, aliases (64 KB)

  const int head = blockIdx.x;
  const int tid = threadIdx.x, lane = tid & 31, wave = tid >> 5;
  const float* Ag = kkinv + (size_t)head * 16384;
  const float* Bg = ktv   + (size_t)head * 16384;

  for (int i = 0; i < 64; ++i) {
    int idx = tid + i * 256;
    int rr = idx >> 7, cc = idx & 127;
    smA[idx]            = (bf16)Ag[idx];    // inv[rr][cc]
    smB[cc * 128 + rr]  = (bf16)Bg[idx];    // ktv^T[cc][rr]
  }
  __syncthreads();

  v8f acc[8];
#pragma unroll
  for (int e = 0; e < 8; ++e) acc[e] = zero_v8f();
  for (int k0 = 0; k0 < 128; k0 += 32) {
    FragU afrag = load_frag<128>(smA, wave * 16 + (lane & 15), k0, lane);
#pragma unroll
    for (int e = 0; e < 8; ++e) {
      FragU bfrag = load_frag<128>(smB, e * 16 + (lane & 15), k0, lane);
      acc[e] = wmma_bf16(afrag, bfrag, acc[e]);
    }
  }
  __syncthreads();   // done reading smA/smB; safe to alias as smP

  const int hi16 = lane >> 4, nc = lane & 15;
#pragma unroll
  for (int e = 0; e < 8; ++e) {
    int col = e * 16 + nc;
#pragma unroll
    for (int v = 0; v < 8; ++v) {
      int row = wave * 16 + v + hi16 * 8;
      smP[row * 128 + col] = acc[e][v];
    }
  }
  __syncthreads();

  bf16* out = score + (size_t)head * 16384;
  if (tid < 128) {
    float m = -3.4e38f;
    for (int c = 0; c < 128; ++c) m = fmaxf(m, smP[tid * 128 + c]);
    float s = 0.0f;
    for (int c = 0; c < 128; ++c) {
      float e0 = __expf(smP[tid * 128 + c] - m);
      smP[tid * 128 + c] = e0;
      s += e0;
    }
    float rs = 1.0f / s;
    for (int c = 0; c < 128; ++c)
      out[tid * 128 + c] = (bf16)(smP[tid * 128 + c] * rs);
  }
}

// ---------------------------------------------------------------------------
// out = q @ score per head.  Grid (32 row-blocks, 64 heads).
// Q tiles staged via async global->LDS copies (pure bf16 copy-through).
// ---------------------------------------------------------------------------
__global__ __launch_bounds__(256)
void attn_kernel(const bf16* __restrict__ Qb, const bf16* __restrict__ score,
                 bf16* __restrict__ attn) {
  __shared__ __align__(16) bf16 St[128 * 128];   // score^T : St[e][d]   (32 KB)
  __shared__ __align__(16) bf16 Qs[128 * 32];    // Q tile               (8 KB)

  const int head = blockIdx.y;
  const int rb   = blockIdx.x;
  const int b = head / NH_, h = head % NH_;
  const size_t base = (size_t)b * S_ * E_ + (size_t)h * HD_;
  const int row0 = rb * 128;
  const int tid = threadIdx.x, lane = tid & 31, wave = tid >> 5;

  const bf16* sc = score + (size_t)head * 16384;
  for (int i = 0; i < 64; ++i) {
    int idx = tid + i * 256;
    int d = idx >> 7, e = idx & 127;
    St[e * 128 + d] = sc[idx];
  }

  v8f acc[8];
#pragma unroll
  for (int e = 0; e < 8; ++e) acc[e] = zero_v8f();

  for (int k0 = 0; k0 < 128; k0 += 32) {
    // Q tile: 128x32 bf16, straight copy -> async engines.
#pragma unroll
    for (int i = 0; i < 2; ++i) {
      int ci = tid + i * 256;            // 512 chunks of 8 bf16
      int ss = ci >> 2, c = (ci & 3) * 8;
      copy16_g2l(Qb + base + (size_t)(row0 + ss) * E_ + (k0 + c),
                 Qs + ss * 32 + c);
    }
    async_join();
    __syncthreads();
    FragU afrag = load_frag<32>(Qs, wave * 16 + (lane & 15), 0, lane);
#pragma unroll
    for (int e = 0; e < 8; ++e) {
      FragU bfrag = load_frag<128>(St, e * 16 + (lane & 15), k0, lane);
      acc[e] = wmma_bf16(afrag, bfrag, acc[e]);
    }
    __syncthreads();
  }

  const int hi16 = lane >> 4, nc = lane & 15;
#pragma unroll
  for (int e = 0; e < 8; ++e) {
    int col = e * 16 + nc;
#pragma unroll
    for (int v = 0; v < 8; ++v) {
      int row = wave * 16 + v + hi16 * 8;
      attn[base + (size_t)(row0 + row) * E_ + col] = (bf16)acc[e][v];
    }
  }
}

// ---------------------------------------------------------------------------
extern "C" void kernel_launch(void* const* d_in, const int* in_sizes, int n_in,
                              void* d_out, int out_size, void* d_ws, size_t ws_size,
                              hipStream_t stream) {
  (void)in_sizes; (void)n_in; (void)out_size; (void)ws_size;

  const float* x     = (const float*)d_in[0];
  const float* alpha = (const float*)d_in[1];
  const float* Wq    = (const float*)d_in[2];
  const float* bq    = (const float*)d_in[3];
  const float* Wk    = (const float*)d_in[4];
  const float* bk    = (const float*)d_in[5];
  const float* Wv    = (const float*)d_in[6];
  const float* bv    = (const float*)d_in[7];
  const float* Wfc   = (const float*)d_in[8];
  const float* bfc   = (const float*)d_in[9];
  float* out = (float*)d_out;

  // Workspace layout (~202 MB).
  char* ws = (char*)d_ws;
  const size_t QKV_BYTES = (size_t)MTOT * E_ * sizeof(bf16);   // 64 MB each
  bf16*  Qbf  = (bf16*)(ws);
  bf16*  Kbf  = (bf16*)(ws + QKV_BYTES);
  bf16*  Vbf  = (bf16*)(ws + 2 * QKV_BYTES);                   // reused as attn
  float* kk   = (float*)(ws + 3 * QKV_BYTES);                  // 4 MB (-> inverse)
  float* ktv  = (float*)(ws + 3 * QKV_BYTES + (size_t)4194304);
  bf16*  scb  = (bf16*)(ws + 3 * QKV_BYTES + (size_t)8388608); // 2 MB

  dim3 gG(E_ / 128, MTOT / 128);   // (16, 128)
  dim3 blk(256);

  gemm_bias_kernel<true,  false><<<gG, blk, 0, stream>>>(x, Wq, bq, Qbf, MTOT, E_, E_);
  gemm_bias_kernel<true,  false><<<gG, blk, 0, stream>>>(x, Wk, bk, Kbf, MTOT, E_, E_);
  gemm_bias_kernel<true,  false><<<gG, blk, 0, stream>>>(x, Wv, bv, Vbf, MTOT, E_, E_);

  kkt_ktv_kernel<<<64, blk, 0, stream>>>(Kbf, Vbf, alpha, kk, ktv);
  invert_kernel<<<64, blk, 0, stream>>>(kk);
  score_kernel<<<64, blk, 65536, stream>>>(kk, ktv, scb);

  attn_kernel<<<dim3(32, 64), blk, 0, stream>>>(Qbf, scb, Vbf /* reuse as attn */);

  gemm_bias_kernel<false, true><<<gG, blk, 0, stream>>>(Vbf, Wfc, bfc, out, MTOT, E_, E_);
}
